// Attention_55087250538754
// MI455X (gfx1250) — matrix operand order repaired and checked by reading, not compiled, not run on hardware
//
#include <hip/hip_runtime.h>
#include <hip/hip_bf16.h>
#include <math.h>

// ---------------------------------------------------------------------------
// Types
// ---------------------------------------------------------------------------
typedef _Float16 h16_t;
typedef __attribute__((ext_vector_type(16))) _Float16 v16h;
typedef __attribute__((ext_vector_type(8)))  _Float16 v8h;
typedef __attribute__((ext_vector_type(8)))  float    v8f;

union F16Frag { v16h v; v8h h[2]; _Float16 e[16]; };

static constexpr int Dm    = 512;
static constexpr int Bb    = 4;
static constexpr int Ss    = 4096;
static constexpr int Mrows = Bb * Ss;                  // 16384
static constexpr float QSCALE = 0.04419417382415922f;  // 1/sqrt(512)

__device__ __forceinline__ int lane_id() { return (int)(threadIdx.x & 31); }

// A-fragment (16x32 f16, row-major source, leading dim ld in elements)
__device__ __forceinline__ v16h load_a_frag(const h16_t* tile, int ld) {
    int lane = lane_id();
    int row  = lane & 15;
    int koff = (lane >> 4) * 8;
    const h16_t* p = tile + (size_t)row * ld + koff;
    F16Frag f;
    f.h[0] = *(const v8h*)(p);
    f.h[1] = *(const v8h*)(p + 16);
    return f.v;
}

// B-fragment (32x16 f16). Element (k,n) = W[n, k] for row-major W[N,K].
__device__ __forceinline__ v16h load_b_frag(const h16_t* base, int ld) {
    int lane = lane_id();
    int n  = lane & 15;
    int kk = (lane >> 4) * 8;
    const h16_t* p = base + (size_t)n * ld + kk;
    F16Frag f;
    f.h[0] = *(const v8h*)(p);
    f.h[1] = *(const v8h*)(p + 16);
    return f.v;
}

// CDNA5 async global -> LDS copy, 16 bytes per lane (ASYNCcnt tracked)
__device__ __forceinline__ void async_ld16(unsigned lds_off, unsigned long long gaddr) {
    asm volatile("global_load_async_to_lds_b128 %0, %1, off"
                 :: "v"(lds_off), "v"(gaddr) : "memory");
}

// ---------------------------------------------------------------------------
// Kernel: f32 -> f16 convert (weights)
// ---------------------------------------------------------------------------
__global__ void cvt_f32_f16(const float* __restrict__ src, h16_t* __restrict__ dst, int n) {
    int i = blockIdx.x * blockDim.x + threadIdx.x;
    if (i < n) dst[i] = (h16_t)src[i];
}

// ---------------------------------------------------------------------------
// Kernel: LayerNorm (one 256-thread block per row of 512)
// ---------------------------------------------------------------------------
__global__ __launch_bounds__(256) void ln_kernel(const float* __restrict__ x,
                                                 const float* __restrict__ g,
                                                 const float* __restrict__ b,
                                                 h16_t* __restrict__ out) {
    int row = blockIdx.x;
    int t   = threadIdx.x;
    const float* xr = x + (size_t)row * Dm;
    float v0 = xr[t], v1 = xr[t + 256];
    float s1 = v0 + v1;
    float s2 = v0 * v0 + v1 * v1;
    #pragma unroll
    for (int off = 16; off >= 1; off >>= 1) {
        s1 += __shfl_xor(s1, off, 32);
        s2 += __shfl_xor(s2, off, 32);
    }
    __shared__ float red[2][8];
    int w = t >> 5;
    if ((t & 31) == 0) { red[0][w] = s1; red[1][w] = s2; }
    __syncthreads();
    float ts1 = 0.f, ts2 = 0.f;
    #pragma unroll
    for (int i = 0; i < 8; ++i) { ts1 += red[0][i]; ts2 += red[1][i]; }
    float mu   = ts1 * (1.0f / Dm);
    float var  = ts2 * (1.0f / Dm) - mu * mu;
    float rstd = rsqrtf(var + 1e-5f);
    h16_t* o = out + (size_t)row * Dm;
    o[t]       = (h16_t)((v0 - mu) * rstd * g[t] + b[t]);
    o[t + 256] = (h16_t)((v1 - mu) * rstd * g[t + 256] + b[t + 256]);
}

// ---------------------------------------------------------------------------
// Kernel: generic f16 WMMA GEMM  C[M,N] = A[M,K] @ W[N,K]^T + bias, *scale
// block = 256 (8 waves), wave -> 16(M) x 64(N).  grid = (M/128, N/64)
// ---------------------------------------------------------------------------
template <bool F16OUT>
__global__ __launch_bounds__(256) void gemm_wmma(const h16_t* __restrict__ A,
                                                 const h16_t* __restrict__ W,
                                                 const float* __restrict__ bias,
                                                 h16_t* __restrict__ outH,
                                                 float* __restrict__ outF,
                                                 int K, int Nld, float scale) {
    int w    = (int)(threadIdx.x >> 5);
    int lane = lane_id();
    int m0   = blockIdx.x * 128 + w * 16;
    int n0   = blockIdx.y * 64;

    v8f acc[4];
    #pragma unroll
    for (int j = 0; j < 4; ++j)
        #pragma unroll
        for (int r = 0; r < 8; ++r) acc[j][r] = 0.f;

    for (int k0 = 0; k0 < K; k0 += 32) {
        v16h a = load_a_frag(A + (size_t)m0 * K + k0, K);
        #pragma unroll
        for (int j = 0; j < 4; ++j) {
            v16h bfr = load_b_frag(W + (size_t)(n0 + 16 * j) * K + k0, K);
            acc[j] = __builtin_amdgcn_wmma_f32_16x16x32_f16(
                false, a, false, bfr, (short)0, acc[j], false, false);
        }
    }

    int rowh = (lane >> 4) << 3;
    int col  = lane & 15;
    #pragma unroll
    for (int j = 0; j < 4; ++j) {
        #pragma unroll
        for (int r = 0; r < 8; ++r) {
            int   cc  = n0 + 16 * j + col;
            float val = (acc[j][r] + bias[cc]) * scale;
            size_t idx = (size_t)(m0 + rowh + r) * Nld + cc;
            if (F16OUT) outH[idx] = (h16_t)val;
            else        outF[idx] = val;
        }
    }
}

// ---------------------------------------------------------------------------
// Kernel: RoPE (interleaved, theta=10000) on q,k; pass-through v. f16 in/out.
// ---------------------------------------------------------------------------
__global__ __launch_bounds__(256) void rope_kernel(const h16_t* __restrict__ qkv,
                                                   h16_t* __restrict__ qo,
                                                   h16_t* __restrict__ ko,
                                                   h16_t* __restrict__ vo) {
    int row = blockIdx.x;
    int t   = threadIdx.x;                 // pair index 0..255
    int pos = row & (Ss - 1);
    float inv = __powf(10000.f, -((float)(2 * t)) / (float)Dm);
    float ang = (float)pos * inv;
    float c = __cosf(ang), s = __sinf(ang);
    const h16_t* r = qkv + (size_t)row * (3 * Dm);
    float q0 = (float)r[2 * t],          q1 = (float)r[2 * t + 1];
    float k0 = (float)r[Dm + 2 * t],     k1 = (float)r[Dm + 2 * t + 1];
    float v0 = (float)r[2 * Dm + 2 * t], v1 = (float)r[2 * Dm + 2 * t + 1];
    size_t o = (size_t)row * Dm + 2 * t;
    qo[o]     = (h16_t)(q0 * c - q1 * s);
    qo[o + 1] = (h16_t)(q1 * c + q0 * s);
    ko[o]     = (h16_t)(k0 * c - k1 * s);
    ko[o + 1] = (h16_t)(k1 * c + k0 * s);
    vo[o]     = (h16_t)v0;
    vo[o + 1] = (h16_t)v1;
}

// ---------------------------------------------------------------------------
// Kernel: V transpose  v[B*S,512] -> vT[B,512,S]  (f16, 32x32 LDS tiles)
// ---------------------------------------------------------------------------
__global__ void vtrans_kernel(const h16_t* __restrict__ v, h16_t* __restrict__ vT) {
    __shared__ h16_t tile[32][33];
    int s0 = blockIdx.x * 32, c0 = blockIdx.y * 32, b = blockIdx.z;
    int tx = threadIdx.x, ty = threadIdx.y;
    #pragma unroll
    for (int i = ty; i < 32; i += 8)
        tile[i][tx] = v[(size_t)(b * Ss + s0 + i) * Dm + c0 + tx];
    __syncthreads();
    #pragma unroll
    for (int i = ty; i < 32; i += 8)
        vT[((size_t)b * Dm + c0 + i) * Ss + s0 + tx] = tile[tx][i];
}

// ---------------------------------------------------------------------------
// Flash attention with CDNA5 async global->LDS double-buffered staging.
// block = 256 thr (8 waves); wave -> 16 query rows x 256 output channels.
// grid = (S/64, B)
// ---------------------------------------------------------------------------

// Stage one 32-key K tile (32KB, contiguous) + V tile (512 chan x 32 keys, 32KB)
// 16 async b128 ops per thread, tracked on ASYNCcnt.
__device__ __forceinline__ void stage_kv(const char* kg, const char* vg, int j0,
                                         h16_t* kdst, h16_t* vdst, int tid) {
    unsigned kl = (unsigned)(uintptr_t)(void*)kdst;
    unsigned vl = (unsigned)(uintptr_t)(void*)vdst;
    const char* ksrc = kg + (size_t)j0 * (Dm * 2);   // 32 contiguous rows of K
    const char* vsrc = vg + (size_t)j0 * 2;          // key-column offset in vT rows
    #pragma unroll
    for (int i = 0; i < 8; ++i) {
        unsigned off = (unsigned)(i * 256 + tid) * 16;
        async_ld16(kl + off, (unsigned long long)(uintptr_t)(ksrc + off));
    }
    #pragma unroll
    for (int i = 0; i < 8; ++i) {
        int idx  = i * 256 + tid;                    // 0..2047
        int chan = idx >> 2;
        int part = (idx & 3) * 16;
        async_ld16(vl + (unsigned)idx * 16,
                   (unsigned long long)(uintptr_t)(vsrc + (size_t)chan * (Ss * 2) + part));
    }
}

__global__ __launch_bounds__(256) void flash_kernel(const h16_t* __restrict__ Q,
                                                    const h16_t* __restrict__ Kv,
                                                    const h16_t* __restrict__ VT,
                                                    h16_t* __restrict__ O) {
    __shared__ h16_t qsm[64 * 512];        // 64 KB: 64 query rows, reused all blocks
    __shared__ h16_t ksm[2][32 * 512];     // 2 x 32 KB: K tile double buffer
    __shared__ h16_t vsm[2][512 * 32];     // 2 x 32 KB: V^T tile double buffer
    __shared__ h16_t pbuf[8][16 * 32];     // 8 KB: per-wave P staging

    int tid   = (int)threadIdx.x;
    int w     = tid >> 5;
    int lane  = tid & 31;
    int half  = w & 1;                      // which 256 output channels
    int qt    = w >> 1;                     // query tile within block (0..3)
    int m0    = blockIdx.x * 64;
    int batch = blockIdx.y;

    const char* qg = (const char*)(Q  + (size_t)(batch * Ss + m0) * Dm);
    const char* kg = (const char*)(Kv + (size_t)batch * Ss * Dm);
    const char* vg = (const char*)(VT + (size_t)batch * Dm * Ss);

    // ---- stage Q block (64 KB contiguous) + first KV tile, async ----
    {
        unsigned ql = (unsigned)(uintptr_t)(void*)qsm;
        #pragma unroll
        for (int i = 0; i < 16; ++i) {
            unsigned off = (unsigned)(i * 256 + tid) * 16;
            async_ld16(ql + off, (unsigned long long)(uintptr_t)(qg + off));
        }
    }
    stage_kv(kg, vg, 0, ksm[0], vsm[0], tid);

    float mrow[8], lrow[8];
    #pragma unroll
    for (int r = 0; r < 8; ++r) { mrow[r] = -1e30f; lrow[r] = 0.f; }

    v8f o[16];
    #pragma unroll
    for (int t = 0; t < 16; ++t)
        #pragma unroll
        for (int r = 0; r < 8; ++r) o[t][r] = 0.f;

    int rowh = (lane >> 4) << 3;
    int col  = lane & 15;
    const h16_t* qtile = qsm + (size_t)qt * 16 * Dm;
    h16_t*       pb    = pbuf[w];

    for (int j0 = 0; j0 < Ss; j0 += 32) {
        int cur = (j0 >> 5) & 1;
        // issue next tile's DMA, then retire current tile (ASYNCcnt is in-order)
        if (j0 + 32 < Ss) {
            stage_kv(kg, vg, j0 + 32, ksm[cur ^ 1], vsm[cur ^ 1], tid);
            asm volatile("s_wait_asynccnt 16" ::: "memory");
        } else {
            asm volatile("s_wait_asynccnt 0" ::: "memory");
        }
        __syncthreads();   // current tile visible to all waves

        const h16_t* kt = ksm[cur];
        const h16_t* vt = vsm[cur];

        // ---- scores S = q @ k^T  (q pre-scaled by 1/sqrt(d)) ----
        v8f s0t, s1t;
        #pragma unroll
        for (int r = 0; r < 8; ++r) { s0t[r] = 0.f; s1t[r] = 0.f; }
        for (int k0 = 0; k0 < Dm; k0 += 32) {
            v16h a  = load_a_frag(qtile + k0, Dm);
            v16h b0 = load_b_frag(kt + k0, Dm);
            v16h b1 = load_b_frag(kt + (size_t)16 * Dm + k0, Dm);
            s0t = __builtin_amdgcn_wmma_f32_16x16x32_f16(false, a, false, b0, (short)0, s0t, false, false);
            s1t = __builtin_amdgcn_wmma_f32_16x16x32_f16(false, a, false, b1, (short)0, s1t, false, false);
        }

        // ---- online softmax update (rows live in 16-lane groups) ----
        #pragma unroll
        for (int r = 0; r < 8; ++r) {
            float v0 = s0t[r], v1 = s1t[r];
            float mx = fmaxf(v0, v1);
            mx = fmaxf(mx, __shfl_xor(mx, 1, 32));
            mx = fmaxf(mx, __shfl_xor(mx, 2, 32));
            mx = fmaxf(mx, __shfl_xor(mx, 4, 32));
            mx = fmaxf(mx, __shfl_xor(mx, 8, 32));
            float mnew  = fmaxf(mrow[r], mx);
            float alpha = __expf(mrow[r] - mnew);
            float p0 = __expf(v0 - mnew);
            float p1 = __expf(v1 - mnew);
            float ls = p0 + p1;
            ls += __shfl_xor(ls, 1, 32);
            ls += __shfl_xor(ls, 2, 32);
            ls += __shfl_xor(ls, 4, 32);
            ls += __shfl_xor(ls, 8, 32);
            lrow[r] = lrow[r] * alpha + ls;
            mrow[r] = mnew;
            #pragma unroll
            for (int t = 0; t < 16; ++t) o[t][r] *= alpha;
            int rl = (r + rowh) * 32 + col;
            pb[rl]      = (h16_t)p0;
            pb[rl + 16] = (h16_t)p1;
        }
        asm volatile("s_wait_dscnt 0" ::: "memory");   // P staged in LDS

        // ---- O += P @ V  (both fragments from LDS) ----
        {
            int prow = lane & 15;
            int kk   = (lane >> 4) * 8;
            F16Frag pf;
            pf.h[0] = *(const v8h*)&pb[prow * 32 + kk];
            pf.h[1] = *(const v8h*)&pb[prow * 32 + kk + 16];
            #pragma unroll
            for (int t = 0; t < 16; ++t) {
                v16h bv = load_b_frag(vt + (size_t)(half * 256 + t * 16) * 32, 32);
                o[t] = __builtin_amdgcn_wmma_f32_16x16x32_f16(
                    false, pf.v, false, bv, (short)0, o[t], false, false);
            }
        }
        __syncthreads();   // all waves done with buf[cur] before DMA overwrites
    }

    // ---- normalize and store ----
    #pragma unroll
    for (int t = 0; t < 16; ++t) {
        #pragma unroll
        for (int r = 0; r < 8; ++r) {
            float val = o[t][r] / lrow[r];
            size_t idx = (size_t)(batch * Ss + m0 + qt * 16 + rowh + r) * Dm
                       + half * 256 + t * 16 + col;
            O[idx] = (h16_t)val;
        }
    }
}

// ---------------------------------------------------------------------------
// Host: kernel_launch
// ---------------------------------------------------------------------------
extern "C" void kernel_launch(void* const* d_in, const int* in_sizes, int n_in,
                              void* d_out, int out_size, void* d_ws, size_t ws_size,
                              hipStream_t stream) {
    const float* x     = (const float*)d_in[0];
    const float* ln_g  = (const float*)d_in[1];
    const float* ln_b  = (const float*)d_in[2];
    const float* qkv_w = (const float*)d_in[3];
    const float* qkv_b = (const float*)d_in[4];
    const float* in_w  = (const float*)d_in[5];
    const float* in_b  = (const float*)d_in[6];
    const float* out_w = (const float*)d_in[7];
    const float* out_b = (const float*)d_in[8];
    float* out = (float*)d_out;

    char* ws = (char*)d_ws;
    h16_t* qkvw16 = (h16_t*)(ws + 0);                         // 1,572,864 B
    h16_t* inw16  = (h16_t*)(ws + 1572864);                   // 1,572,864 B
    h16_t* outw16 = (h16_t*)(ws + 3145728);                   // 524,288 B
    h16_t* h16    = (h16_t*)(ws + 3670016);                   // 16 MB (reused for o16)
    h16_t* qkv16  = (h16_t*)(ws + 20447232);                  // 50 MB (reused: q2,k2,v2)
    h16_t* qr     = (h16_t*)(ws + 70778880);                  // 16 MB (reused for vT)
    h16_t* kr     = (h16_t*)(ws + 87556096);                  // 16 MB
    h16_t* vv     = (h16_t*)(ws + 104333312);                 // 16 MB  (end ~121 MB)
    h16_t* q2  = qkv16;
    h16_t* k2  = qkv16 + (size_t)Mrows * Dm;
    h16_t* v2  = qkv16 + (size_t)2 * Mrows * Dm;
    h16_t* vT  = qr;
    h16_t* o16 = h16;

    // 1) weights f32 -> f16
    cvt_f32_f16<<<3072, 256, 0, stream>>>(qkv_w, qkvw16, 3 * Dm * Dm);
    cvt_f32_f16<<<3072, 256, 0, stream>>>(in_w,  inw16,  3 * Dm * Dm);
    cvt_f32_f16<<<1024, 256, 0, stream>>>(out_w, outw16, Dm * Dm);

    // 2) LayerNorm -> h (f16)
    ln_kernel<<<Mrows, 256, 0, stream>>>(x, ln_g, ln_b, h16);

    // 3) qkv = h @ qkv_w^T + qkv_b  (f16 out)
    gemm_wmma<true><<<dim3(Mrows / 128, (3 * Dm) / 64), 256, 0, stream>>>(
        h16, qkvw16, qkv_b, qkv16, nullptr, Dm, 3 * Dm, 1.0f);

    // 4) RoPE on q,k; copy v
    rope_kernel<<<Mrows, 256, 0, stream>>>(qkv16, qr, kr, vv);

    // 5) in_proj: q (folds 1/sqrt(d)), k, v
    gemm_wmma<true><<<dim3(Mrows / 128, Dm / 64), 256, 0, stream>>>(
        qr, inw16, in_b, q2, nullptr, Dm, Dm, QSCALE);
    gemm_wmma<true><<<dim3(Mrows / 128, Dm / 64), 256, 0, stream>>>(
        kr, inw16 + (size_t)Dm * Dm, in_b + Dm, k2, nullptr, Dm, Dm, 1.0f);
    gemm_wmma<true><<<dim3(Mrows / 128, Dm / 64), 256, 0, stream>>>(
        vv, inw16 + (size_t)2 * Dm * Dm, in_b + 2 * Dm, v2, nullptr, Dm, Dm, 1.0f);

    // 6) transpose V for contiguous B-fragments in P@V
    vtrans_kernel<<<dim3(Ss / 32, Dm / 32, Bb), dim3(32, 8), 0, stream>>>(v2, vT);

    // 7) flash attention -> o (f16)
    flash_kernel<<<dim3(Ss / 64, Bb), 256, 0, stream>>>(q2, k2, vT, o16);

    // 8) out_proj -> f32 d_out
    gemm_wmma<false><<<dim3(Mrows / 128, Dm / 64), 256, 0, stream>>>(
        o16, outw16, out_b, nullptr, out, Dm, Dm, 1.0f);
}
